// DRXNet_17214228922616
// MI455X (gfx1250) — matrix operand order repaired and checked by reading, not compiled, not run on hardware
//
#include <hip/hip_runtime.h>
#include <hip/hip_bf16.h>

typedef __attribute__((ext_vector_type(16))) _Float16 v16h;
typedef __attribute__((ext_vector_type(8)))  float    v8f;

#define NCRY 2048
#define NNODE 16384

struct HeadP   { const float* bg1; const float* wg2; const float* bm1; const float* bm2; const _Float16* pk; };
struct CryHeadP{ const float* bg1; const float* wg1; const float* wg2; const float* bm1; const float* wm1; const float* bm2; const float* wm2; };
struct FinalP {
  const float *re1b,*re1w,*re2b,*re2w;   // rate_emb
  const float *ce1b,*ce1w,*ce2b,*ce2w;   // cycle_emb
  const float *erfb,*erfw,*ergb,*ergw;   // encode_rate fc / gate
  const float *ecfb,*ecfw,*ecgb,*ecgw;   // encode_cycle fc / gate
  const float *dNw;                      // delta_N_w
  const float *ev1b,*ev1w,*ev2b,*ev2w;   // enc_vol
  const float *avb,*avw;                 // add_vol.fc
  const float *f1b,*f1w,*f2b,*f2w;       // fc.fc1 / fc.fc2
};

__device__ __forceinline__ float softplusf(float x) {
  return fmaxf(x, 0.f) + log1pf(expf(-fabsf(x)));
}
__device__ __forceinline__ float sigmoidf(float x) {
  return 1.f / (1.f + expf(-x));
}
__device__ __forceinline__ float wave_allsum(float v) {
  for (int o = 16; o; o >>= 1) v += __shfl_xor(v, o);
  return v;
}

// ---------------------------------------------------------------------------
// Pack a row-major f32 weight matrix (din = 32*ktiles, dout) into f16 WMMA
// B-operand tiles: tile T=(kt*ntiles+nt), per lane L: n = nt*16 + (L&15),
// k = kt*32 + (jh<8 ? 0 : 16) + (L>>4)*8 + (jh&7)  (slot jh = 2*j + h), 16 halfs per lane.
// ---------------------------------------------------------------------------
__global__ void pack_b_kernel(const float* __restrict__ W, _Float16* __restrict__ dst,
                              int dout, int ntiles) {
  int T = blockIdx.x;
  int kt = T / ntiles, nt = T % ntiles;
  int L = threadIdx.x;
  int n = nt * 16 + (L & 15);
  int g = L >> 4;
  _Float16* d = dst + T * 512 + L * 16;
#pragma unroll
  for (int jh = 0; jh < 16; ++jh) {
    int j = jh >> 1, h = jh & 1;
    int k = kt * 32 + ((jh < 8) ? 0 : 16) + g * 8 + (jh & 7);
    d[jh] = (_Float16)W[k * dout + n];
  }
}

// ---------------------------------------------------------------------------
// Embedding: fea[n,0:31] = elem_fea[n,:] @ W(200x31) + b ; fea[n,31] = weight
// One wave per node (lane = output column).
// ---------------------------------------------------------------------------
__global__ __launch_bounds__(256) void embed_kernel(const float* __restrict__ ef,
                                                    const float* __restrict__ W,
                                                    const float* __restrict__ b,
                                                    const float* __restrict__ ew,
                                                    float* __restrict__ fea0) {
  int node = blockIdx.x * 8 + (threadIdx.x >> 5);
  int t = threadIdx.x & 31;
  if (t < 31) {
    float acc = b[t];
    const float* row = ef + (size_t)node * 200;
    for (int k = 0; k < 200; ++k) acc += row[k] * W[k * 31 + t];
    fea0[node * 32 + t] = acc;
  } else {
    fea0[node * 32 + 31] = ew[node];
  }
}

// ---------------------------------------------------------------------------
// One fused graph layer: 1 workgroup (8 waves) per crystal of 8 nodes / 64 edges.
// cat (64x64) -> fused [gate fc1 + gate.out] and [msg fc1 -> LDS] sharing the
// A-tiles (4 WMMAs/iteration), segment softmax over 8 edges/node, msg fc2 via
// WMMA, weighted pool with head-mean + residual.
// ---------------------------------------------------------------------------
__global__ __launch_bounds__(256) void graph_layer_kernel(const float* __restrict__ fea_in,
                                                          float* __restrict__ fea_out,
                                                          const float* __restrict__ elem_w,
                                                          HeadP H0, HeadP H1, HeadP H2) {
  __shared__ float s_fea[8][32];
  __shared__ float s_w[8];
  __shared__ _Float16 s_catA[2][4][32][16] __attribute__((aligned(32))); // [kt][mt][lane][16]
  __shared__ _Float16 s_hidA[8][4][32][16] __attribute__((aligned(32))); // msg hidden, A-packed
  __shared__ float s_gate[64];
  __shared__ float s_coeff[64];
  __shared__ float s_acc[8][32];

  const int c = blockIdx.x;
  const int tid = threadIdx.x;
  const int lane = tid & 31;
  const int wave = tid >> 5;

  {
    float v = fea_in[c * 256 + tid];
    s_fea[tid >> 5][tid & 31] = v;
    s_acc[tid >> 5][tid & 31] = v;   // residual baked into accumulator
  }
  if (tid < 8) s_w[tid] = elem_w[c * 8 + tid];
  __syncthreads();

  // Build cat = [fea[self], fea[nbr]] directly in WMMA A-operand layout.
  for (int p = tid; p < 4096; p += 256) {
    int jh = p & 15, j = jh >> 1, h = jh & 1;
    int L = (p >> 4) & 31;
    int mt = (p >> 9) & 3;
    int kt = p >> 11;
    int m = mt * 16 + (L & 15);
    int g = L >> 4;
    int kk = ((j < 4) ? 0 : 16) + g * 8 + (j & 3) * 2 + h;
    int k = kt * 32 + kk;
    int i = m >> 3, jn = m & 7;
    float v = (k < 32) ? s_fea[i][k] : s_fea[jn][k - 32];
    s_catA[kt][mt][L][jh] = (_Float16)v;
  }
  __syncthreads();

  for (int hd = 0; hd < 3; ++hd) {
    const HeadP hp = (hd == 0) ? H0 : (hd == 1 ? H1 : H2);
    if (tid < 64) s_gate[tid] = 0.f;
    __syncthreads();

    const int mofs = (lane >> 4) << 3;
    const int nlo = lane & 15;

    // ---- fused gate fc1 + msg fc1 over shared cat A-tiles ----
#pragma unroll 2
    for (int q = 0; q < 8; ++q) {
      int T = wave * 8 + q;
      int mt = T & 3, nt = T >> 2;
      v16h a0 = *(const v16h*)(&s_catA[0][mt][lane][0]);
      v16h a1 = *(const v16h*)(&s_catA[1][mt][lane][0]);
      const _Float16* pg = hp.pk;
      const _Float16* pm = hp.pk + 16384;
      v16h bg0 = *(const v16h*)(pg + (0 * 16 + nt) * 512 + lane * 16);
      v16h bg1 = *(const v16h*)(pg + (1 * 16 + nt) * 512 + lane * 16);
      v16h bm0 = *(const v16h*)(pm + (0 * 16 + nt) * 512 + lane * 16);
      v16h bm1 = *(const v16h*)(pm + (1 * 16 + nt) * 512 + lane * 16);
      v8f ga = {};
      v8f ma = {};
      ga = __builtin_amdgcn_wmma_f32_16x16x32_f16(false, a0, false, bg0, (short)0, ga, false, false);
      ma = __builtin_amdgcn_wmma_f32_16x16x32_f16(false, a0, false, bm0, (short)0, ma, false, false);
      ga = __builtin_amdgcn_wmma_f32_16x16x32_f16(false, a1, false, bg1, (short)0, ga, false, false);
      ma = __builtin_amdgcn_wmma_f32_16x16x32_f16(false, a1, false, bm1, (short)0, ma, false, false);

      int n = nt * 16 + nlo;

      // gate: bias + relu + gate.out weight, half-wave reduce, LDS accumulate
      {
        float bb = hp.bg1[n];
        float w2 = hp.wg2[n];
#pragma unroll
        for (int r = 0; r < 8; ++r) {
          float v = fmaxf(ga[r] + bb, 0.f) * w2;
          v += __shfl_xor(v, 1, 16);
          v += __shfl_xor(v, 2, 16);
          v += __shfl_xor(v, 4, 16);
          v += __shfl_xor(v, 8, 16);
          if (nlo == 0) atomicAdd(&s_gate[mt * 16 + r + mofs], v);
        }
      }
      // msg: bias + relu, scatter into A-packed hidden layout
      {
        float bb = hp.bm1[n];
        int kt2 = n >> 5;
        int kk = n & 31;
        int g2 = (kk >> 3) & 1;
        int j2 = ((kk & 7) >> 1) + ((kk >> 4) << 2);
        int h2 = kk & 1;
#pragma unroll
        for (int r = 0; r < 8; ++r) {
          int m = mt * 16 + r + mofs;
          float v = fmaxf(ma[r] + bb, 0.f);
          s_hidA[kt2][mt][(m & 15) + (g2 << 4)][j2 * 2 + h2] = (_Float16)v;
        }
      }
    }
    __syncthreads();

    // ---- segment softmax: each node over its 8 edges, weighted by nbr weight
    if (tid < 8) {
      int i = tid;
      float mx = -1e30f;
      for (int j = 0; j < 8; ++j) mx = fmaxf(mx, s_gate[i * 8 + j]);
      float e[8];
      float sum = 0.f;
      for (int j = 0; j < 8; ++j) { e[j] = s_w[j] * expf(s_gate[i * 8 + j] - mx); sum += e[j]; }
      float inv = 1.f / (sum + 1e-10f);
      for (int j = 0; j < 8; ++j) s_coeff[i * 8 + j] = e[j] * inv;
    }
    __syncthreads();

    // ---- msg fc2 (64x256 @ 256x32) + weighted pooling, 1 tile per wave ----
    {
      int mt = wave & 3, nt = wave >> 2;
      const _Float16* pm2 = hp.pk + 32768;
      v8f acc = {};
#pragma unroll
      for (int kt = 0; kt < 8; ++kt) {
        v16h a = *(const v16h*)(&s_hidA[kt][mt][lane][0]);
        v16h b = *(const v16h*)(pm2 + (kt * 2 + nt) * 512 + lane * 16);
        acc = __builtin_amdgcn_wmma_f32_16x16x32_f16(false, a, false, b, (short)0, acc, false, false);
      }
      int n = nt * 16 + nlo;
      float bb = hp.bm2[n];
#pragma unroll
      for (int r = 0; r < 8; ++r) {
        int m = mt * 16 + r + mofs;
        float val = acc[r] + bb;
        atomicAdd(&s_acc[m >> 3][n], s_coeff[m] * val * (1.f / 3.f));
      }
    }
    __syncthreads();
  }

  fea_out[c * 256 + tid] = s_acc[tid >> 5][tid & 31];
}

// ---------------------------------------------------------------------------
// Crystal pooling: 3 WeightedAttentionPooling heads over 8 nodes/crystal.
// Small GEMMs (K=32), VALU implementation, 1 workgroup per crystal.
// ---------------------------------------------------------------------------
__global__ __launch_bounds__(256) void cry_pool_kernel(const float* __restrict__ fea,
                                                       const float* __restrict__ elem_w,
                                                       float* __restrict__ crys,
                                                       CryHeadP C0, CryHeadP C1, CryHeadP C2) {
  __shared__ float s_x[8][32];
  __shared__ float s_hid[8][256];
  __shared__ float s_gate[8];
  __shared__ float s_coeff[8];
  __shared__ float s_w[8];
  __shared__ float s_out[32];

  int c = blockIdx.x;
  int tid = threadIdx.x;
  s_x[tid >> 5][tid & 31] = fea[c * 256 + tid];
  if (tid < 8) s_w[tid] = elem_w[c * 8 + tid];
  if (tid < 32) s_out[tid] = 0.f;
  __syncthreads();

  for (int hd = 0; hd < 3; ++hd) {
    const CryHeadP hp = (hd == 0) ? C0 : (hd == 1 ? C1 : C2);
    if (tid < 8) s_gate[tid] = 0.f;
    __syncthreads();

    for (int idx = tid; idx < 2048; idx += 256) {
      int node = idx >> 8, hc = idx & 255;
      float ag = hp.bg1[hc], am = hp.bm1[hc];
      for (int k = 0; k < 32; ++k) {
        float xv = s_x[node][k];
        ag += xv * hp.wg1[k * 256 + hc];
        am += xv * hp.wm1[k * 256 + hc];
      }
      s_hid[node][hc] = fmaxf(am, 0.f);
      atomicAdd(&s_gate[node], fmaxf(ag, 0.f) * hp.wg2[hc]);
    }
    __syncthreads();

    if (tid == 0) {
      float mx = -1e30f;
      for (int n = 0; n < 8; ++n) mx = fmaxf(mx, s_gate[n]);
      float e[8];
      float sum = 0.f;
      for (int n = 0; n < 8; ++n) { e[n] = s_w[n] * expf(s_gate[n] - mx); sum += e[n]; }
      float inv = 1.f / (sum + 1e-10f);
      for (int n = 0; n < 8; ++n) s_coeff[n] = e[n] * inv;
    }
    __syncthreads();

    {
      int node = tid >> 5, f = tid & 31;
      float acc = hp.bm2[f];
      for (int k = 0; k < 256; ++k) acc += s_hid[node][k] * hp.wm2[k * 32 + f];
      atomicAdd(&s_out[f], s_coeff[node] * acc * (1.f / 3.f));
    }
    __syncthreads();
  }
  if (tid < 32) crys[c * 32 + tid] = s_out[tid];
}

// ---------------------------------------------------------------------------
// Condition encoders + voltage branch + q_out + analytic dq/dVii.
// One wave per crystal, lane = feature index (F = 32 = wave32).
// ---------------------------------------------------------------------------
__global__ __launch_bounds__(256) void final_kernel(const float* __restrict__ crys,
                                                    const float* __restrict__ Vw,
                                                    const float* __restrict__ rate,
                                                    const float* __restrict__ cycle,
                                                    const float* __restrict__ Vii,
                                                    float* __restrict__ out,
                                                    FinalP P) {
  int wave = threadIdx.x >> 5;
  int f = threadIdx.x & 31;
  int f16i = f & 15;
  int c = blockIdx.x * 8 + wave;

  float cf = crys[c * 32 + f];
  float r = rate[c], cy = cycle[c];
  float v0 = Vw[c * 2], v1 = Vw[c * 2 + 1], vii = Vii[c];

  // rate_emb: 1 -> 32 (relu) -> 16
  float hr = fmaxf(r * P.re1w[f] + P.re1b[f], 0.f);
  float rf = P.re2b[f16i];
  for (int k = 0; k < 32; ++k) rf += __shfl(hr, k) * P.re2w[k * 16 + f16i];

  // cycle_emb
  float hc = fmaxf(cy * P.ce1w[f] + P.ce1b[f], 0.f);
  float cfe = P.ce2b[f16i];
  for (int k = 0; k < 32; ++k) cfe += __shfl(hc, k) * P.ce2w[k * 16 + f16i];

  // encode_rate: z = [crys(32), rate_fea(16)] -> relu(fc)*sigmoid(gate)
  float aF = P.erfb[f], aG = P.ergb[f];
  for (int k = 0; k < 32; ++k) { float zk = __shfl(cf, k); aF += zk * P.erfw[k * 32 + f]; aG += zk * P.ergw[k * 32 + f]; }
  for (int k = 0; k < 16; ++k) { float zk = __shfl(rf, k); aF += zk * P.erfw[(32 + k) * 32 + f]; aG += zk * P.ergw[(32 + k) * 32 + f]; }
  float cond_rate = cf + fmaxf(aF, 0.f) * sigmoidf(aG);

  // encode_cycle
  float bF = P.ecfb[f], bG = P.ecgb[f];
  for (int k = 0; k < 32; ++k) { float zk = __shfl(cond_rate, k); bF += zk * P.ecfw[k * 32 + f]; bG += zk * P.ecgw[k * 32 + f]; }
  for (int k = 0; k < 16; ++k) { float zk = __shfl(cfe, k); bF += zk * P.ecfw[(32 + k) * 32 + f]; bG += zk * P.ecgw[(32 + k) * 32 + f]; }
  float cond_cycle = cond_rate + fmaxf(bF, 0.f) * sigmoidf(bG) * (cy - 1.f) * P.dNw[f];

  // enc_vol: [v0,v1,vii](3) -> 64 (relu) -> 32 ; keep d/dVii
  float preA = P.ev1b[f]      + v0 * P.ev1w[f]      + v1 * P.ev1w[64 + f]      + vii * P.ev1w[128 + f];
  float preB = P.ev1b[32 + f] + v0 * P.ev1w[32 + f] + v1 * P.ev1w[96 + f]      + vii * P.ev1w[160 + f];
  float hA = fmaxf(preA, 0.f), hB = fmaxf(preB, 0.f);
  float dA = (preA > 0.f) ? P.ev1w[128 + f] : 0.f;
  float dB = (preB > 0.f) ? P.ev1w[160 + f] : 0.f;
  float xv = P.ev2b[f], dxv = 0.f;
  for (int k = 0; k < 32; ++k) {
    float w1v = P.ev2w[k * 32 + f];
    float w2v = P.ev2w[(k + 32) * 32 + f];
    xv  += __shfl(hA, k) * w1v + __shfl(hB, k) * w2v;
    dxv += __shfl(dA, k) * w1v + __shfl(dB, k) * w2v;
  }

  // add_vol + output head (softplus chain)
  float u = xv + cond_cycle;
  float a = P.avb[f];
  for (int k = 0; k < 32; ++k) a += __shfl(u, k) * P.avw[k * 32 + f];
  float x  = softplusf(a);
  float sa = sigmoidf(a);
  float h1 = P.f1b[f];
  for (int k = 0; k < 32; ++k) h1 += __shfl(x, k) * P.f1w[k * 32 + f];
  float s1  = softplusf(h1);
  float sh1 = sigmoidf(h1);
  float z = P.f2b[0] + wave_allsum(s1 * P.f2w[f]);
  float q  = softplusf(z);
  float sz = sigmoidf(z);

  // backward: dq/dVii
  float t = sh1 * P.f2w[f];
  float gx = 0.f;
  for (int k = 0; k < 32; ++k) gx += __shfl(t, k) * P.f1w[f * 32 + k];
  float grad = sz * wave_allsum(gx * sa * dxv);

  if (f == 0) {
    out[c * 2]     = q;
    out[c * 2 + 1] = grad;
  }
}

// ---------------------------------------------------------------------------
// Host-side orchestration (all launches on `stream`).
// Param leaves follow JAX pytree (sorted dict key) flattening, base index 9.
// ---------------------------------------------------------------------------
extern "C" void kernel_launch(void* const* d_in, const int* in_sizes, int n_in,
                              void* d_out, int out_size, void* d_ws, size_t ws_size,
                              hipStream_t stream) {
  (void)in_sizes; (void)n_in; (void)out_size; (void)ws_size;
  const float* elem_w   = (const float*)d_in[0];
  const float* elem_fea = (const float*)d_in[1];
  const float* Vw    = (const float*)d_in[5];
  const float* rate  = (const float*)d_in[6];
  const float* cycle = (const float*)d_in[7];
  const float* Vii   = (const float*)d_in[8];
  auto P = [&](int i) { return (const float*)d_in[9 + i]; };

  float* fea0 = (float*)d_ws;                  // 16384*32
  float* fea1 = fea0 + NNODE * 32;             // 16384*32
  float* crys = fea1 + NNODE * 32;             // 2048*32
  _Float16* pk = (_Float16*)(crys + NCRY * 32);

  // pack graph-layer weights into WMMA B-tile layout (f16)
  for (int g = 0; g < 3; ++g)
    for (int h = 0; h < 3; ++h) {
      int base = 49 + 24 * g + 8 * h;
      _Float16* ph = pk + (size_t)(g * 3 + h) * 40960;
      pack_b_kernel<<<32, 32, 0, stream>>>(P(base + 1), ph,          256, 16); // gate.W1 64x256
      pack_b_kernel<<<32, 32, 0, stream>>>(P(base + 5), ph + 16384,  256, 16); // msg.W1  64x256
      pack_b_kernel<<<16, 32, 0, stream>>>(P(base + 7), ph + 32768,   32,  2); // msg.W2 256x32
    }

  // embedding (leaf 31 = embedding.b, 32 = embedding.w)
  embed_kernel<<<NNODE / 8, 256, 0, stream>>>(elem_fea, P(32), P(31), elem_w, fea0);

  // 3 fused graph layers (ping-pong fea0/fea1)
  for (int g = 0; g < 3; ++g) {
    HeadP H[3];
    for (int h = 0; h < 3; ++h) {
      int base = 49 + 24 * g + 8 * h;
      H[h] = { P(base + 0), P(base + 3), P(base + 4), P(base + 6),
               pk + (size_t)(g * 3 + h) * 40960 };
    }
    const float* fin = (g & 1) ? fea1 : fea0;
    float* fout      = (g & 1) ? fea0 : fea1;
    graph_layer_kernel<<<NCRY, 256, 0, stream>>>(fin, fout, elem_w, H[0], H[1], H[2]);
  }

  // crystal pooling (leaves 2..25, 8 per head)
  CryHeadP Ch[3];
  for (int h = 0; h < 3; ++h) {
    int b = 2 + 8 * h;
    Ch[h] = { P(b + 0), P(b + 1), P(b + 3), P(b + 4), P(b + 5), P(b + 6), P(b + 7) };
  }
  cry_pool_kernel<<<NCRY, 256, 0, stream>>>(fea1, elem_w, crys, Ch[0], Ch[1], Ch[2]);

  // condition encoders + voltage branch + gradient
  FinalP FP = {
    P(121), P(122), P(123), P(124),   // rate_emb
    P(26),  P(27),  P(28),  P(29),    // cycle_emb
    P(41),  P(42),  P(43),  P(44),    // encode_rate
    P(37),  P(38),  P(39),  P(40),    // encode_cycle
    P(30),                            // delta_N_w
    P(33),  P(34),  P(35),  P(36),    // enc_vol
    P(0),   P(1),                     // add_vol
    P(45),  P(46),  P(47),  P(48)     // fc
  };
  final_kernel<<<NCRY / 8, 256, 0, stream>>>(crys, Vw, rate, cycle, Vii, (float*)d_out, FP);
}
